// TemporalAttention2_55138790146543
// MI455X (gfx1250) — matrix operand order repaired and checked by reading, not compiled, not run on hardware
//
#include <hip/hip_runtime.h>
#include <cstdint>

#define B_   8
#define T_   1024
#define D_   512
#define WS_  12
#define HALO 11
#define BAND 23
#define NEGV (-1000000000.0f)

typedef __attribute__((ext_vector_type(16))) __bf16 bf16x16;
typedef __attribute__((ext_vector_type(8)))  __bf16 bf16x8;
typedef __attribute__((ext_vector_type(8)))  float  f32x8;

// ---------- bf16 helpers (RNE), avoiding hip_bf16 header dependencies ----------
static __device__ __forceinline__ unsigned short f2bf(float f) {
    union { float f; unsigned u; } v; v.f = f;
    unsigned r = v.u + 0x7FFFu + ((v.u >> 16) & 1u);
    return (unsigned short)(r >> 16);
}
static __device__ __forceinline__ float bf2f(unsigned short h) {
    union { unsigned u; float f; } v; v.u = ((unsigned)h) << 16; return v.f;
}

// ---------- CDNA5 async global->LDS copy (16B per lane) ----------
static __device__ __forceinline__ void async_cp16(unsigned lds_off, const void* gptr) {
    asm volatile("global_load_async_to_lds_b128 %0, %1, off"
                 :: "v"(lds_off), "v"((unsigned long long)(uintptr_t)gptr)
                 : "memory");
}
static __device__ __forceinline__ void wait_async0() {
    asm volatile("s_wait_asynccnt 0x0" ::: "memory");
}

// ---------- WMMA fragment loaders from LDS ----------
// A (16x32 bf16): lane(hs,lm) -> row M=lm, K = hs*16..hs*16+15 (contiguous)
static __device__ __forceinline__ bf16x16 loadA(const unsigned short* p) {
    bf16x8 lo = *(const bf16x8*)p;
    bf16x8 hi = *(const bf16x8*)(p + 8);
    return __builtin_shufflevector(lo, hi, 0,1,2,3,4,5,6,7,8,9,10,11,12,13,14,15);
}
// B (32x16 bf16): lane(hs,lm) -> col N=lm, K = hs*16..hs*16+15 (contiguous)
static __device__ __forceinline__ bf16x16 loadB(const unsigned short* p) {
    bf16x8 lo = *(const bf16x8*)p;
    bf16x8 hi = *(const bf16x8*)(p + 8);
    return __builtin_shufflevector(lo, hi, 0,1,2,3,4,5,6,7,8,9,10,11,12,13,14,15);
}
static __device__ __forceinline__ f32x8 wmma_bf16(bf16x16 a, bf16x16 b, f32x8 c) {
    return __builtin_amdgcn_wmma_f32_16x16x32_bf16(false, a, false, b, (short)0, c, false, false);
}

// ---------- fold conv bias + BN into scale/shift ----------
__global__ void bn_prep_kernel(const float* __restrict__ g, const float* __restrict__ beta,
                               const float* __restrict__ m, const float* __restrict__ v,
                               const float* __restrict__ cb,
                               float* __restrict__ scale, float* __restrict__ shift) {
    int i = blockIdx.x * blockDim.x + threadIdx.x;
    if (i < D_) {
        float s = g[i] * rsqrtf(v[i] + 1e-5f);
        scale[i] = s;
        shift[i] = (cb[i] - m[i]) * s + beta[i];
    }
}

// ---------- (co,ci,kh,kw) f32 -> (co,kh,kw,ci) bf16 ----------
__global__ void reorder_w_kernel(const float* __restrict__ w, unsigned short* __restrict__ o, int KHW) {
    size_t i = (size_t)blockIdx.x * blockDim.x + threadIdx.x;
    size_t total = (size_t)D_ * D_ * KHW;
    if (i >= total) return;
    int k  = (int)(i % KHW);
    size_t r = i / KHW;
    int ci = (int)(r % D_);
    int co = (int)(r / D_);
    o[((size_t)co * KHW + k) * D_ + ci] = f2bf(w[i]);
}

// ---------- banded scores + stable top-12 + gather (bf16 out, im2col layout) ----------
__global__ void topk_gather_kernel(const float* __restrict__ x, unsigned short* __restrict__ g) {
    const int bt = blockIdx.x;
    const int b  = bt / T_;
    const int t  = bt % T_;
    const int wave = threadIdx.x >> 5;
    const int lane = threadIdx.x & 31;

    __shared__ float sc[BAND];
    __shared__ int   ids[WS_];

    const float* xb = x + (size_t)b * T_ * D_;
    const float* xt = xb + (size_t)t * D_;

    if (wave < BAND) {
        int j = t - HALO + wave;
        float s = NEGV;
        if (j >= 0 && j < T_) {
            const float* xj = xb + (size_t)j * D_;
            float acc = 0.f;
            for (int d = lane; d < D_; d += 32) acc += xt[d] * xj[d];
            for (int off = 16; off > 0; off >>= 1) acc += __shfl_xor(acc, off, 32);
            s = acc * 0.04419417382415922f;  // 1/sqrt(512)
        }
        if (lane == 0) sc[wave] = s;
    }
    __syncthreads();

    if (threadIdx.x == 0) {
        int cnt = 0;
        for (int w = 0; w < BAND && cnt < WS_; ++w) {
            float sw = sc[w];
            int rank = 0;
            for (int u = 0; u < BAND; ++u) {
                float su = sc[u];
                if (su > sw || (su == sw && u < w)) ++rank;
            }
            if (rank < WS_) ids[cnt++] = t - HALO + w;
        }
    }
    __syncthreads();

    for (int idx = threadIdx.x; idx < WS_ * D_; idx += blockDim.x) {
        int w = idx / D_, d = idx - w * D_;
        int j = ids[w];
        g[((size_t)(b * T_ + t) * WS_ + w) * D_ + d] = f2bf(xb[(size_t)j * D_ + d]);
    }
}

// ---------- implicit-GEMM conv + fused BN + ReLU (WMMA bf16 -> f32) ----------
// Block tile: 64 co x 256 n.  8 waves = 2 co-groups(32) x 4 n-groups(64).
// Wave tile : 32 co x 64 n = 8 accumulators; 2 A frags x 4 B frags -> 8 WMMA/step.
// Double-buffered LDS; async global->LDS copies overlap WMMA compute.
// in  : [B][H][W_IN][512] bf16   (ci contiguous)
// wgt : [512][KH][KW][512] bf16  (ci contiguous)
// out : [B*H*W_OUT][512]  (bf16 or f32), position-major
template<int KH, int KW, int W_IN, int W_OUT, int H, bool OUT_BF16>
__global__ __launch_bounds__(256)
void conv_bn_relu_kernel(const unsigned short* __restrict__ in,
                         const unsigned short* __restrict__ wgt,
                         const float* __restrict__ scale,
                         const float* __restrict__ shift,
                         void* __restrict__ outv) {
    constexpr int CI = D_;
    constexpr int KSTEPS = KH * KW * (CI / 32);   // K-chunks of 32

    __shared__ alignas(16) unsigned short As[2][64][32];    //  8 KB
    __shared__ alignas(16) unsigned short Bs[2][256][32];   // 32 KB

    const int tid  = threadIdx.x;
    const int wave = tid >> 5, lane = tid & 31;
    const int coSub = wave & 1;    // 2 co groups of 32
    const int nSub  = wave >> 1;   // 4 n groups of 64
    const int lm = lane & 15, hs = lane >> 4;

    const int nTile0  = blockIdx.x * 256;
    const int coTile0 = blockIdx.y * 64;

    // A cooperative load: 64 rows x 4x16B quarters (one b128 per thread)
    const int aRowI = tid >> 2;
    const int aQ    = tid & 3;
    const unsigned short* aBase =
        wgt + (size_t)(coTile0 + aRowI) * (KH * KW * CI) + aQ * 8;

    // B cooperative load: row n = nTile0 + tid, 64B (four b128 per thread)
    const int n   = nTile0 + tid;
    const int bI  = n / (H * W_OUT);
    const int rem = n - bI * (H * W_OUT);
    const int hO  = rem / W_OUT;
    const int wO  = rem - hO * W_OUT;

    auto issue = [&](int step, int buf) {
        const int kseg = step >> 4;           // CI/32 == 16 chunks per (kh,kw)
        const int ck   = (step & 15) << 5;
        const int kh   = kseg / KW;
        const int kw   = kseg - kh * KW;
        // A panel
        async_cp16((unsigned)(uintptr_t)&As[buf][aRowI][aQ * 8],
                   aBase + (size_t)(kh * KW + kw) * CI + ck);
        // B panel (zero-fill rows that fall in the h-padding)
        const int hIn = hO + kh - 2;
        if (hIn >= 0 && hIn < H) {
            const unsigned short* src =
                in + ((size_t)((bI * H + hIn) * W_IN + (wO + kw))) * CI + ck;
            const unsigned l0 = (unsigned)(uintptr_t)&Bs[buf][tid][0];
            async_cp16(l0,      src);
            async_cp16(l0 + 16, src + 8);
            async_cp16(l0 + 32, src + 16);
            async_cp16(l0 + 48, src + 24);
        } else {
            uint4 z = {0u, 0u, 0u, 0u};
            uint4* d = (uint4*)&Bs[buf][tid][0];
            d[0] = z; d[1] = z; d[2] = z; d[3] = z;
        }
    };

    f32x8 acc[2][4] = {};

    // prologue: fill buffer 0
    issue(0, 0);
    wait_async0();
    __syncthreads();

    for (int step = 0; step < KSTEPS; ++step) {
        const int buf = step & 1;
        if (step + 1 < KSTEPS) issue(step + 1, buf ^ 1);

        // fragments for this step
        bf16x16 a0 = loadA(&As[buf][coSub * 32      + lm][hs * 16]);
        bf16x16 a1 = loadA(&As[buf][coSub * 32 + 16 + lm][hs * 16]);
        bf16x16 b0 = loadB(&Bs[buf][nSub * 64      + lm][hs * 16]);
        bf16x16 b1 = loadB(&Bs[buf][nSub * 64 + 16 + lm][hs * 16]);
        bf16x16 b2 = loadB(&Bs[buf][nSub * 64 + 32 + lm][hs * 16]);
        bf16x16 b3 = loadB(&Bs[buf][nSub * 64 + 48 + lm][hs * 16]);

        acc[0][0] = wmma_bf16(a0, b0, acc[0][0]);
        acc[0][1] = wmma_bf16(a0, b1, acc[0][1]);
        acc[0][2] = wmma_bf16(a0, b2, acc[0][2]);
        acc[0][3] = wmma_bf16(a0, b3, acc[0][3]);
        acc[1][0] = wmma_bf16(a1, b0, acc[1][0]);
        acc[1][1] = wmma_bf16(a1, b1, acc[1][1]);
        acc[1][2] = wmma_bf16(a1, b2, acc[1][2]);
        acc[1][3] = wmma_bf16(a1, b3, acc[1][3]);

        // publish next buffer; also fences reads of this buffer before
        // it is overwritten by the next iteration's issue
        wait_async0();
        __syncthreads();
    }

    // epilogue: C/D layout -> VGPR r holds (M = hs*8 + r, N = lm)
#pragma unroll
    for (int i2 = 0; i2 < 2; ++i2) {
#pragma unroll
        for (int j = 0; j < 4; ++j) {
            f32x8 c = acc[i2][j];
            const int nIdx = nTile0 + nSub * 64 + j * 16 + lm;
#pragma unroll
            for (int r = 0; r < 8; ++r) {
                const int co = coTile0 + coSub * 32 + i2 * 16 + hs * 8 + r;
                float v = c[r] * scale[co] + shift[co];
                v = fmaxf(v, 0.f);
                if (OUT_BF16)
                    ((unsigned short*)outv)[(size_t)nIdx * D_ + co] = f2bf(v);
                else
                    ((float*)outv)[(size_t)nIdx * D_ + co] = v;
            }
        }
    }
}

// ---------- 2x2 max pool: (B,1024,8,512)bf16 -> (B,512,4,512)bf16 ----------
__global__ void pool1_kernel(const unsigned short* __restrict__ in, unsigned short* __restrict__ out) {
    size_t i = (size_t)blockIdx.x * blockDim.x + threadIdx.x;
    const size_t total = (size_t)B_ * 512 * 4 * D_;
    if (i >= total) return;
    int co = (int)(i & 511); size_t r = i >> 9;
    int w2 = (int)(r & 3); r >>= 2;
    int h2 = (int)(r & 511);
    int b  = (int)(r >> 9);
    int h = 2 * h2, w = 2 * w2;
    const size_t s0 = ((size_t)((b * T_ + h) * 8 + w)) * D_ + co;
    float m0 = fmaxf(bf2f(in[s0]),            bf2f(in[s0 + D_]));
    float m1 = fmaxf(bf2f(in[s0 + 8 * D_]),   bf2f(in[s0 + 9 * D_]));
    out[i] = f2bf(fmaxf(m0, m1));
}

// ---------- 2x2 max pool: (B,512,2,512)f32 -> d_out (B,512co,256) f32 ----------
__global__ void pool2_kernel(const float* __restrict__ in, float* __restrict__ out) {
    size_t i = (size_t)blockIdx.x * blockDim.x + threadIdx.x;
    const size_t total = (size_t)B_ * D_ * 256;
    if (i >= total) return;
    int h2 = (int)(i & 255); size_t r = i >> 8;
    int co = (int)(r & 511);
    int b  = (int)(r >> 9);
    const size_t s0 = ((size_t)((b * 512 + 2 * h2) * 2)) * D_ + co;
    float m0 = fmaxf(in[s0],            in[s0 + D_]);
    float m1 = fmaxf(in[s0 + 2 * D_],   in[s0 + 3 * D_]);
    out[((size_t)(b * D_ + co)) * 256 + h2] = fmaxf(m0, m1);
}

extern "C" void kernel_launch(void* const* d_in, const int* in_sizes, int n_in,
                              void* d_out, int out_size, void* d_ws, size_t ws_size,
                              hipStream_t stream) {
    (void)in_sizes; (void)n_in; (void)out_size; (void)ws_size;

    const float* x       = (const float*)d_in[0];
    const float* conv1_w = (const float*)d_in[1];
    const float* conv1_b = (const float*)d_in[2];
    const float* bn1_g   = (const float*)d_in[3];
    const float* bn1_b   = (const float*)d_in[4];
    const float* bn1_m   = (const float*)d_in[5];
    const float* bn1_v   = (const float*)d_in[6];
    const float* conv2_w = (const float*)d_in[7];
    const float* conv2_b = (const float*)d_in[8];
    const float* bn2_g   = (const float*)d_in[9];
    const float* bn2_b   = (const float*)d_in[10];
    const float* bn2_m   = (const float*)d_in[11];
    const float* bn2_v   = (const float*)d_in[12];

    char* p = (char*)d_ws;
    auto take = [&](size_t bytes) -> void* {
        void* r = (void*)p;
        p += (bytes + 255) & ~(size_t)255;
        return r;
    };
    unsigned short* g    = (unsigned short*)take((size_t)B_ * T_ * WS_ * D_ * 2);   // 96 MB
    unsigned short* w1b  = (unsigned short*)take((size_t)D_ * 25 * D_ * 2);          // 12.5 MB
    unsigned short* w2b  = (unsigned short*)take((size_t)D_ * 15 * D_ * 2);          // 7.5 MB
    float* s1 = (float*)take(D_ * 4);
    float* t1 = (float*)take(D_ * 4);
    float* s2 = (float*)take(D_ * 4);
    float* t2 = (float*)take(D_ * 4);
    unsigned short* tmp1 = (unsigned short*)take((size_t)B_ * T_ * 8 * D_ * 2);      // 64 MB
    unsigned short* p1   = (unsigned short*)take((size_t)B_ * 512 * 4 * D_ * 2);     // 16 MB
    float*          tmp2 = (float*)take((size_t)B_ * 512 * 2 * D_ * 4);              // 16 MB

    // 1) fold BN + conv bias
    bn_prep_kernel<<<2, 256, 0, stream>>>(bn1_g, bn1_b, bn1_m, bn1_v, conv1_b, s1, t1);
    bn_prep_kernel<<<2, 256, 0, stream>>>(bn2_g, bn2_b, bn2_m, bn2_v, conv2_b, s2, t2);

    // 2) weight reorder/convert
    reorder_w_kernel<<<(unsigned)(((size_t)D_ * D_ * 25 + 255) / 256), 256, 0, stream>>>(conv1_w, w1b, 25);
    reorder_w_kernel<<<(unsigned)(((size_t)D_ * D_ * 15 + 255) / 256), 256, 0, stream>>>(conv2_w, w2b, 15);

    // 3) banded attention top-k + gather (23 waves per (b,t))
    topk_gather_kernel<<<B_ * T_, BAND * 32, 0, stream>>>(x, g);

    // 4) conv1 + BN1 + ReLU : N = 8*1024*8 = 65536 positions
    {
        dim3 grid((B_ * T_ * 8) / 256, D_ / 64);
        conv_bn_relu_kernel<5, 5, 12, 8, T_, true>
            <<<grid, 256, 0, stream>>>(g, w1b, s1, t1, tmp1);
    }

    // 5) pool 2x2 -> (B,512,4,512)
    pool1_kernel<<<(unsigned)(((size_t)B_ * 512 * 4 * D_) / 256), 256, 0, stream>>>(tmp1, p1);

    // 6) conv2 + BN2 + ReLU : N = 8*512*2 = 8192 positions
    {
        dim3 grid((B_ * 512 * 2) / 256, D_ / 64);
        conv_bn_relu_kernel<5, 3, 4, 2, 512, false>
            <<<grid, 256, 0, stream>>>(p1, w2b, s2, t2, tmp2);
    }

    // 7) pool 2x2 -> d_out (B, 512, 256) f32
    pool2_kernel<<<(unsigned)(((size_t)B_ * D_ * 256) / 256), 256, 0, stream>>>(tmp2, (float*)d_out);
}
